// GQA_47218870452983
// MI455X (gfx1250) — matrix operand order repaired and checked by reading, not compiled, not run on hardware
//
#include <hip/hip_runtime.h>
#include <hip/hip_bf16.h>

typedef __attribute__((ext_vector_type(16))) __bf16 v16bf;
typedef __attribute__((ext_vector_type(8)))  float  v8f;

// ---------------------------------------------------------------- helpers
__device__ __forceinline__ unsigned short f2bf(float f) {
    unsigned int u = __float_as_uint(f);
    u += 0x7fffu + ((u >> 16) & 1u);        // round-to-nearest-even
    return (unsigned short)(u >> 16);
}

__device__ __forceinline__ v8f zero8() {
    v8f z;
#pragma unroll
    for (int i = 0; i < 8; ++i) z[i] = 0.0f;
    return z;
}

union FragBF {
    v16bf v;
    uint4 q[2];
};

// A fragment (16x32 bf16, M x K): lane L -> row (L&15),
// K = {0..7,16..23} / {8..15,24..31} per lane half.
__device__ __forceinline__ FragBF load_frag_a(const unsigned short* __restrict__ base,
                                              int ld, int row0, int k0) {
    const int lane = threadIdx.x & 31;
    const unsigned short* p =
        base + (size_t)(row0 + (lane & 15)) * ld + k0 + ((lane >> 4) << 3);
    FragBF f;
    f.q[0] = *(const uint4*)(p);
    f.q[1] = *(const uint4*)(p + 16);
    return f;
}

// raw-pointer fragment loads (LDS tiles; p already points at lane's row/k)
__device__ __forceinline__ FragBF frag_raw_a(const unsigned short* p) {
    FragBF f;
    f.q[0] = *(const uint4*)(p);
    f.q[1] = *(const uint4*)(p + 16);
    return f;
}
__device__ __forceinline__ FragBF frag_raw_b(const unsigned short* p) {
    FragBF f;
    f.q[0] = *(const uint4*)(p);
    f.q[1] = *(const uint4*)(p + 16);
    return f;
}

__device__ __forceinline__ v8f wmma_bf16(v16bf a, v16bf b, v8f c) {
    return __builtin_amdgcn_wmma_f32_16x16x32_bf16(
        false, a, false, b, (short)0, c, false, false);
}

// async 16-byte global -> LDS copy (per-lane addresses, tracked by ASYNCcnt)
__device__ __forceinline__ void async_ld16(unsigned lds_addr, const void* gaddr) {
    asm volatile("global_load_async_to_lds_b128 %0, %1, off"
                 :: "v"(lds_addr), "v"((unsigned long long)(uintptr_t)gaddr)
                 : "memory");
}
#define WAIT_ASYNC8() asm volatile("s_wait_asynccnt 8" ::: "memory")
#define WAIT_ASYNC0() asm volatile("s_wait_asynccnt 0" ::: "memory")

// ---------------------------------------------------------------- fp32 -> bf16
__global__ __launch_bounds__(256) void cvt_x_bf16(const float4* __restrict__ in,
                                                  unsigned short* __restrict__ out,
                                                  int n4) {
    int i = blockIdx.x * 256 + threadIdx.x;
    if (i >= n4) return;
    float4 v = in[i];
    uint2 o;
    o.x = (unsigned)f2bf(v.x) | ((unsigned)f2bf(v.y) << 16);
    o.y = (unsigned)f2bf(v.z) | ((unsigned)f2bf(v.w) << 16);
    *(uint2*)(out + (size_t)i * 4) = o;
}

// W[K,N] fp32 -> Wt[N,K] bf16 (tiled transpose through LDS)
__global__ __launch_bounds__(256) void cvt_w_t(const float* __restrict__ W,
                                               unsigned short* __restrict__ Wt,
                                               int K, int N) {
    __shared__ float tile[32][33];
    const int nb = blockIdx.x * 32, kb = blockIdx.y * 32;
    const int tx = threadIdx.x & 31;
    const int ty = threadIdx.x >> 5;   // 0..7
#pragma unroll
    for (int r = 0; r < 4; ++r) {
        int kk = ty + r * 8;
        tile[kk][tx] = W[(size_t)(kb + kk) * N + nb + tx];
    }
    __syncthreads();
#pragma unroll
    for (int r = 0; r < 4; ++r) {
        int nn = ty + r * 8;
        Wt[(size_t)(nb + nn) * K + kb + tx] = f2bf(tile[tx][nn]);
    }
}

// ---------------------------------------------------------------- GEMM
// C[M,N] = (A[M,K] * Bt[N,K]^T + bias[N]) * scaleOut
// Block tile 128x128, 4 waves (2x2), 64x64 per wave.
// A/B 128x32 tiles double-buffered in LDS via GLOBAL_LOAD_ASYNC_TO_LDS_B128.
// Rows padded to 40 bf16 so b128 fragment reads are bank-conflict-free.
// MODE 0: bf16 row-major; MODE 1: bf16 -> Vt[b][kv][dh][s]; MODE 2: f32.
template <int MODE>
__global__ __launch_bounds__(128) void gemm_bf16(
    const unsigned short* __restrict__ A,
    const unsigned short* __restrict__ Bt,
    const float* __restrict__ bias,
    void* __restrict__ Cout,
    int M, int N, int K, float scaleOut) {
    __shared__ __align__(16) unsigned short Atile[2][128][40];
    __shared__ __align__(16) unsigned short Btile[2][128][40];

    const int lane = threadIdx.x & 31;
    const int w = threadIdx.x >> 5;
    const int tid = threadIdx.x;
    const int mBlock = blockIdx.y * 128;
    const int nBlock = blockIdx.x * 128;
    const int mW = (w & 1) * 64;
    const int nW = (w >> 1) * 64;
    const int hi = lane >> 4, lo = lane & 15;

    auto stage = [&](int buf, int ksrc) {
#pragma unroll
        for (int j = 0; j < 4; ++j) {
            const int c = tid + 128 * j;            // 0..511
            const int r = c >> 2, off = (c & 3) * 8;
            async_ld16((unsigned)(uintptr_t)&Atile[buf][r][off],
                       A + (size_t)(mBlock + r) * K + ksrc + off);
        }
#pragma unroll
        for (int j = 0; j < 4; ++j) {
            const int c = tid + 128 * j;
            const int r = c >> 2, off = (c & 3) * 8;
            async_ld16((unsigned)(uintptr_t)&Btile[buf][r][off],
                       Bt + (size_t)(nBlock + r) * K + ksrc + off);
        }
    };

    v8f acc[4][4];
#pragma unroll
    for (int mt = 0; mt < 4; ++mt)
#pragma unroll
        for (int nt = 0; nt < 4; ++nt) acc[mt][nt] = zero8();

    stage(0, 0);
    stage(1, 32);

    for (int k = 0; k < K; k += 32) {
        const int cur = (k >> 5) & 1;
        if (k + 32 < K) { WAIT_ASYNC8(); } else { WAIT_ASYNC0(); }
        __syncthreads();

        FragBF a[4];
#pragma unroll
        for (int mt = 0; mt < 4; ++mt)
            a[mt] = frag_raw_a(&Atile[cur][mW + mt * 16 + lo][hi * 8]);
#pragma unroll
        for (int nt = 0; nt < 4; ++nt) {
            FragBF bfr = frag_raw_b(&Btile[cur][nW + nt * 16 + lo][hi * 8]);
#pragma unroll
            for (int mt = 0; mt < 4; ++mt)
                acc[mt][nt] = wmma_bf16(a[mt].v, bfr.v, acc[mt][nt]);
        }

        __syncthreads();
        if (k + 64 < K) stage(cur, k + 64);
    }

#pragma unroll
    for (int nt = 0; nt < 4; ++nt) {
        const int n = nBlock + nW + nt * 16 + lo;
        const float bv = bias ? bias[n] : 0.0f;
#pragma unroll
        for (int mt = 0; mt < 4; ++mt) {
#pragma unroll
            for (int i = 0; i < 8; ++i) {
                const int m = mBlock + mW + mt * 16 + i + hi * 8;
                const float val = (acc[mt][nt][i] + bv) * scaleOut;
                if (MODE == 0) {
                    ((unsigned short*)Cout)[(size_t)m * N + n] = f2bf(val);
                } else if (MODE == 1) {
                    const int b  = m >> 11;        // S = 2048
                    const int s  = m & 2047;
                    const int kv = n >> 7;         // DH = 128
                    const int dh = n & 127;
                    ((unsigned short*)Cout)[(((size_t)(b * 4 + kv) * 128 + dh) << 11) + s] =
                        f2bf(val);
                } else {
                    ((float*)Cout)[(size_t)m * N + n] = val;
                }
            }
        }
    }
}

// ---------------------------------------------------------------- attention
// Block = 4 waves, same (b,h), 4 consecutive 16-row q tiles -> identical K/V
// stream, staged ONCE per block into double-buffered LDS via async copies.
// Row-sum of P via WMMA against all-ones B fragment; row-max via butterflies.
__global__ __launch_bounds__(128) void attn_fwd(
    const unsigned short* __restrict__ Qb,   // [B*S, 2048] (Q pre-scaled)
    const unsigned short* __restrict__ Kb,   // [B*S, 512]
    const unsigned short* __restrict__ Vt,   // [B*KV, 128, 2048]
    unsigned short* __restrict__ O) {        // [B*S, 2048]
    const float LOG2E = 1.4426950408889634f;
    const int lane = threadIdx.x & 31;
    const int w = threadIdx.x >> 5;
    const int tid = threadIdx.x;
    const int wg = blockIdx.x * 4 + w;
    const int qTile = wg & 127;            // S/16
    const int h = (wg >> 7) & 15;
    const int b = wg >> 11;
    const int kv = h >> 2;
    const int qBase = b * 2048 + qTile * 16;
    const int kRow0 = b * 2048;
    const int hi = lane >> 4, lo = lane & 15;

    __shared__ __align__(16) unsigned short Ktile[2][32][136];   // [kpos][dh]
    __shared__ __align__(16) unsigned short Vtile[2][128][40];   // [dh][kpos]
    __shared__ __align__(16) unsigned short pT[4][16][40];

    const unsigned short* Vh = Vt + (size_t)(b * 4 + kv) * 128 * 2048;

    auto stage = [&](int buf, int ksrc) {
#pragma unroll
        for (int j = 0; j < 4; ++j) {
            const int c = tid + 128 * j;            // 0..511
            const int r = c >> 4, off = (c & 15) * 8;
            async_ld16((unsigned)(uintptr_t)&Ktile[buf][r][off],
                       Kb + (size_t)(kRow0 + ksrc + r) * 512 + kv * 128 + off);
        }
#pragma unroll
        for (int j = 0; j < 4; ++j) {
            const int c = tid + 128 * j;
            const int r = c >> 2, off = (c & 3) * 8;
            async_ld16((unsigned)(uintptr_t)&Vtile[buf][r][off],
                       Vh + (size_t)r * 2048 + ksrc + off);
        }
    };

    // all-ones B fragment (bf16 1.0 = 0x3F80) for row-sum WMMA
    FragBF onesf;
    onesf.q[0] = make_uint4(0x3f803f80u, 0x3f803f80u, 0x3f803f80u, 0x3f803f80u);
    onesf.q[1] = onesf.q[0];

    // resident Q fragments (16 x 128 = 4 A-fragments)
    FragBF qf[4];
#pragma unroll
    for (int t = 0; t < 4; ++t)
        qf[t] = load_frag_a(Qb, 2048, qBase, h * 128 + t * 32);

    v8f acc[8];
#pragma unroll
    for (int t = 0; t < 8; ++t) acc[t] = zero8();
    v8f lacc = zero8();
    float m8[8];
#pragma unroll
    for (int i = 0; i < 8; ++i) m8[i] = -1e30f;

    stage(0, 0);
    stage(1, 32);

    for (int kb = 0; kb < 2048; kb += 32) {
        const int cur = (kb >> 5) & 1;
        if (kb + 32 < 2048) { WAIT_ASYNC8(); } else { WAIT_ASYNC0(); }
        __syncthreads();

        // scores: 16q x 32kpos, contraction over DH=128
        v8f s0 = zero8(), s1 = zero8();
#pragma unroll
        for (int t = 0; t < 4; ++t) {
            FragBF kf0 = frag_raw_b(&Ktile[cur][lo][t * 32 + hi * 8]);
            s0 = wmma_bf16(qf[t].v, kf0.v, s0);
            FragBF kf1 = frag_raw_b(&Ktile[cur][16 + lo][t * 32 + hi * 8]);
            s1 = wmma_bf16(qf[t].v, kf1.v, s1);
        }

        // online softmax per row (rows split across 16-lane halves)
        float alpha[8];
#pragma unroll
        for (int i = 0; i < 8; ++i) {
            float mx = fmaxf(s0[i], s1[i]);
            mx = fmaxf(mx, __shfl_xor(mx, 1));
            mx = fmaxf(mx, __shfl_xor(mx, 2));
            mx = fmaxf(mx, __shfl_xor(mx, 4));
            mx = fmaxf(mx, __shfl_xor(mx, 8));
            const float mn = fmaxf(m8[i], mx);
            alpha[i] = exp2f((m8[i] - mn) * LOG2E);
            m8[i] = mn;
            const float p0 = exp2f((s0[i] - mn) * LOG2E);
            const float p1 = exp2f((s1[i] - mn) * LOG2E);
            const int row = i + hi * 8;
            pT[w][row][lo]      = f2bf(p0);
            pT[w][row][lo + 16] = f2bf(p1);
        }

        // re-load P in A-fragment layout (same wave: DS ops are in-order)
        FragBF pf = frag_raw_a(&pT[w][lo][hi * 8]);

        // rescale running output + row-sums, then accumulate P*V and P*ones
#pragma unroll
        for (int t = 0; t < 8; ++t)
#pragma unroll
            for (int i = 0; i < 8; ++i) acc[t][i] *= alpha[i];
#pragma unroll
        for (int i = 0; i < 8; ++i) lacc[i] *= alpha[i];
        lacc = wmma_bf16(pf.v, onesf.v, lacc);
#pragma unroll
        for (int t = 0; t < 8; ++t) {
            FragBF vf = frag_raw_b(&Vtile[cur][t * 16 + lo][hi * 8]);
            acc[t] = wmma_bf16(pf.v, vf.v, acc[t]);
        }

        __syncthreads();
        if (kb + 64 < 2048) stage(cur, kb + 64);
    }

    // normalize + store
#pragma unroll
    for (int i = 0; i < 8; ++i) {
        const float inv = 1.0f / lacc[i];
        const int row = qBase + i + hi * 8;
        const size_t o = (size_t)row * 2048 + h * 128 + lo;
#pragma unroll
        for (int t = 0; t < 8; ++t)
            O[o + t * 16] = f2bf(acc[t][i] * inv);
    }
}

// ---------------------------------------------------------------- launch
extern "C" void kernel_launch(void* const* d_in, const int* in_sizes, int n_in,
                              void* d_out, int out_size, void* d_ws, size_t ws_size,
                              hipStream_t stream) {
    const float* x  = (const float*)d_in[0];
    const float* Wq = (const float*)d_in[1];
    const float* bq = (const float*)d_in[2];
    const float* Wk = (const float*)d_in[3];
    const float* bk = (const float*)d_in[4];
    const float* Wv = (const float*)d_in[5];
    const float* bv = (const float*)d_in[6];
    const float* Wo = (const float*)d_in[7];
    const float* bo = (const float*)d_in[8];

    // workspace carve-up (bf16 elements)
    unsigned short* p = (unsigned short*)d_ws;
    unsigned short* xB  = p; p += 8388608;   // [4096,2048]
    unsigned short* WqT = p; p += 4194304;   // [2048,2048]
    unsigned short* WkT = p; p += 1048576;   // [512,2048]
    unsigned short* WvT = p; p += 1048576;   // [512,2048]
    unsigned short* WoT = p; p += 4194304;   // [2048,2048]
    unsigned short* Qb  = p; p += 8388608;   // [4096,2048]
    unsigned short* Kb  = p; p += 2097152;   // [4096,512]
    unsigned short* VtW = p; p += 2097152;   // [2,4,128,2048]
    unsigned short* Ob  = p; p += 8388608;   // [4096,2048]

    // 1. conversions
    cvt_x_bf16<<<(8388608 / 4 + 255) / 256, 256, 0, stream>>>(
        (const float4*)x, xB, 8388608 / 4);
    cvt_w_t<<<dim3(2048 / 32, 2048 / 32), 256, 0, stream>>>(Wq, WqT, 2048, 2048);
    cvt_w_t<<<dim3(512 / 32, 2048 / 32), 256, 0, stream>>>(Wk, WkT, 2048, 512);
    cvt_w_t<<<dim3(512 / 32, 2048 / 32), 256, 0, stream>>>(Wv, WvT, 2048, 512);
    cvt_w_t<<<dim3(2048 / 32, 2048 / 32), 256, 0, stream>>>(Wo, WoT, 2048, 2048);

    // 2. projections (Q pre-scaled by 1/sqrt(DH))
    gemm_bf16<0><<<dim3(2048 / 128, 4096 / 128), 128, 0, stream>>>(
        xB, WqT, bq, Qb, 4096, 2048, 2048, 0.08838834764831845f);
    gemm_bf16<0><<<dim3(512 / 128, 4096 / 128), 128, 0, stream>>>(
        xB, WkT, bk, Kb, 4096, 512, 2048, 1.0f);
    gemm_bf16<1><<<dim3(512 / 128, 4096 / 128), 128, 0, stream>>>(
        xB, WvT, bv, VtW, 4096, 512, 2048, 1.0f);

    // 3. attention: 2(B)*16(H)*128(qTiles) waves / 4 per block
    attn_fwd<<<1024, 128, 0, stream>>>(Qb, Kb, VtW, Ob);

    // 4. output projection -> f32
    gemm_bf16<2><<<dim3(2048 / 128, 4096 / 128), 128, 0, stream>>>(
        Ob, WoT, bo, d_out, 4096, 2048, 2048, 1.0f);
}
